// EncodeLayer_68736656605903
// MI455X (gfx1250) — compile-verified, matrix operand order checked by reading, not run on hardware
//
#include <hip/hip_runtime.h>
#include <hip/hip_bf16.h>
#include <stdint.h>

// ---------------------------------------------------------------------------
// Types for CDNA5 WMMA (wave32). 16x16x32 bf16 -> f32 accumulate.
// ---------------------------------------------------------------------------
typedef __attribute__((ext_vector_type(16))) __bf16 v16bf;
typedef __attribute__((ext_vector_type(8)))  float  v8f;

union FragAB { v16bf v; uint4 u4[2]; unsigned short h[16]; };
union FragC  { v8f v; float f[8]; };

__device__ __forceinline__ unsigned short f2bf(float f) {
  uint32_t u = __float_as_uint(f);
  uint32_t r = (u + 0x7FFFu + ((u >> 16) & 1u)) >> 16;   // RNE
  return (unsigned short)r;
}
__device__ __forceinline__ float bf2f(unsigned short h) {
  return __uint_as_float(((uint32_t)h) << 16);
}
// Orderable-uint mapping so uint atomicMin == float min (exact, deterministic).
__device__ __forceinline__ unsigned f2key(float f) {
  unsigned u = __float_as_uint(f);
  return (u & 0x80000000u) ? ~u : (u | 0x80000000u);
}
__device__ __forceinline__ float key2f(unsigned k) {
  unsigned u = (k & 0x80000000u) ? (k & 0x7FFFFFFFu) : ~k;
  return __uint_as_float(u);
}

// ---------------------------------------------------------------------------
// CDNA5 async global->LDS copy (ASYNCcnt path). LDS dest is the wave-relative
// LDS byte address = low 32 bits of the generic pointer to a __shared__ array
// (ISA: flat/async LDS addressing truncates addr[31:0]).
// ---------------------------------------------------------------------------
__device__ __forceinline__ void async_copy_b128(const void* gaddr, void* ldsp) {
#if defined(__gfx1250__)
  unsigned ldsoff = (unsigned)(unsigned long long)ldsp;
  unsigned long long ga = (unsigned long long)gaddr;
  asm volatile("global_load_async_to_lds_b128 %0, %1, off"
               :: "v"(ldsoff), "v"(ga) : "memory");
#else
  (void)gaddr; (void)ldsp;
#endif
}
__device__ __forceinline__ void wait_asynccnt0() {
#if defined(__gfx1250__)
  asm volatile("s_wait_asynccnt 0" ::: "memory");
#endif
}

// ---------------------------------------------------------------------------
// fp32 -> bf16 elementwise convert (weights)
// ---------------------------------------------------------------------------
__global__ void cvt_f32_to_bf16(const float* __restrict__ src,
                                unsigned short* __restrict__ dst, int n) {
  int i = blockIdx.x * blockDim.x + threadIdx.x;
  if (i < n) dst[i] = f2bf(src[i]);
}

__global__ void init_minkey(unsigned* mk) { *mk = 0xFFFFFFFFu; }

// ---------------------------------------------------------------------------
// Generic C = scale * A(MxK) * W(NxK)^T [+ Res], WMMA bf16.
// Block tile 128x128, 8 waves (4 along M x 2 along N), wave tile 32x64.
// A staged through double-buffered LDS: async-to-LDS when A is bf16,
// convert-through-VGPR when A is fp32.
// ---------------------------------------------------------------------------
template<bool A_BF16, bool OUT_BF16, bool ADD_RES>
__global__ __launch_bounds__(256)
void gemm_tn_wmma(const void* __restrict__ Aptr,
                  const unsigned short* __restrict__ W,   // bf16, N x K row-major
                  void* __restrict__ Outp,
                  const float* __restrict__ Res,
                  int M, int N, int K, float scale) {
  constexpr int KC = 32;
  __shared__ unsigned short As[2][128 * 40];  // 128 rows x 32 halfs, stride 40 (80B)

  const int t    = threadIdx.x;
  const int lane = t & 31;
  const int wid  = t >> 5;
  const int wm   = wid & 3;                 // M sub-tile: rows wm*32
  const int wn   = wid >> 2;                // N sub-tile: cols wn*64
  const int m0   = blockIdx.y * 128;
  const int n0   = blockIdx.x * 128;
  const int l15  = lane & 15;
  const int kh   = (lane >> 4) << 3;        // 0 or 8 (A/B fragment K sub-base)
  const int moff = (lane & 16) ? 8 : 0;

  FragC acc[2][4];
  for (int i = 0; i < 2; ++i)
    for (int j = 0; j < 4; ++j)
      for (int r = 0; r < 8; ++r) acc[i][j].f[r] = 0.f;

  const float* Af          = (const float*)Aptr;
  const unsigned short* Ah = (const unsigned short*)Aptr;

  // Stage one 128x32 A tile (as bf16) into LDS buffer p.
  auto stageA = [&](int p, int kt) {
    if constexpr (A_BF16) {
      #pragma unroll
      for (int i = 0; i < 2; ++i) {
        int fl  = t + 256 * i;              // 8-half chunk id (512 total)
        int row = fl >> 2;
        int c8  = (fl & 3) << 3;
        async_copy_b128(Ah + (size_t)(m0 + row) * K + kt + c8,
                        &As[p][row * 40 + c8]);
      }
    } else {
      #pragma unroll
      for (int i = 0; i < 4; ++i) {
        int fl  = t + 256 * i;              // float4 chunk id (1024 total)
        int row = fl >> 3;
        int c4  = (fl & 7) << 2;
        float4 v = *(const float4*)(Af + (size_t)(m0 + row) * K + kt + c4);
        uint32_t p0 = (uint32_t)f2bf(v.x) | ((uint32_t)f2bf(v.y) << 16);
        uint32_t p1 = (uint32_t)f2bf(v.z) | ((uint32_t)f2bf(v.w) << 16);
        *(uint2*)&As[p][row * 40 + c4] = make_uint2(p0, p1);
      }
    }
  };

  stageA(0, 0);
  if constexpr (A_BF16) wait_asynccnt0();
  __syncthreads();

  int p = 0;
  for (int kt = 0; kt < K; kt += KC) {
    // --- prefetch/stage next tile into the other buffer while computing ---
    if (kt + KC < K) {
      stageA(p ^ 1, kt + KC);
      __builtin_prefetch(W + (size_t)(n0 + wn * 64 + l15) * K + kt + KC, 0, 1);
    }

    // --- B fragments straight from global (weights, L2-resident) ---
    FragAB bfr[4];
    #pragma unroll
    for (int g = 0; g < 4; ++g) {
      int n = n0 + wn * 64 + g * 16 + l15;
      const unsigned short* bp = W + (size_t)n * K + kt + kh;
      bfr[g].u4[0] = *(const uint4*)(bp);
      bfr[g].u4[1] = *(const uint4*)(bp + 16);
    }
    // --- A fragments from LDS buffer p ---
    FragAB afr[2];
    #pragma unroll
    for (int f = 0; f < 2; ++f) {
      const unsigned short* ap = &As[p][(wm * 32 + f * 16 + l15) * 40 + kh];
      afr[f].u4[0] = *(const uint4*)(ap);
      afr[f].u4[1] = *(const uint4*)(ap + 16);
    }
    #pragma unroll
    for (int f = 0; f < 2; ++f)
      #pragma unroll
      for (int g = 0; g < 4; ++g)
        acc[f][g].v = __builtin_amdgcn_wmma_f32_16x16x32_bf16(
            false, afr[f].v, false, bfr[g].v, (short)0, acc[f][g].v, false, false);

    if constexpr (A_BF16) wait_asynccnt0();   // async writes of buf p^1 done
    __syncthreads();                          // everyone done reading buf p
    p ^= 1;
  }

  // --- epilogue ---
  #pragma unroll
  for (int f = 0; f < 2; ++f) {
    int mbase = m0 + wm * 32 + f * 16 + moff;
    #pragma unroll
    for (int g = 0; g < 4; ++g) {
      int n = n0 + wn * 64 + g * 16 + l15;
      #pragma unroll
      for (int r = 0; r < 8; ++r) {
        size_t off = (size_t)(mbase + r) * N + n;
        float val = acc[f][g].f[r] * scale;
        if constexpr (OUT_BF16) {
          ((unsigned short*)Outp)[off] = f2bf(val);
        } else {
          if constexpr (ADD_RES) val += Res[off];
          ((float*)Outp)[off] = val;
        }
      }
    }
  }
}

// ---------------------------------------------------------------------------
// Attention phase 1: scores[b,h,qi,ki] = Qp[b*8+qi, h*64:] . Kp[b*8+ki, h*64:]
// (1/sqrt(dk) folded into Qp). Also global min via orderable-uint atomicMin.
// ---------------------------------------------------------------------------
__global__ __launch_bounds__(256)
void attn_scores(const unsigned short* __restrict__ Qp,
                 const unsigned short* __restrict__ Kp,
                 float* __restrict__ scores, unsigned* __restrict__ minkey) {
  __shared__ unsigned smin;
  if (threadIdx.x == 0) smin = 0xFFFFFFFFu;
  __syncthreads();
  int idx = blockIdx.x * 256 + threadIdx.x;          // < 4096*16*64
  int ki = idx & 7, qi = (idx >> 3) & 7, h = (idx >> 6) & 15, b = idx >> 10;
  const unsigned short* qr = Qp + (size_t)(b * 8 + qi) * 1024 + h * 64;
  const unsigned short* kr = Kp + (size_t)(b * 8 + ki) * 1024 + h * 64;
  float s = 0.f;
  #pragma unroll
  for (int j = 0; j < 64; j += 8) {
    uint4 qa = *(const uint4*)(qr + j);
    uint4 ka = *(const uint4*)(kr + j);
    const unsigned short* qh = (const unsigned short*)&qa;
    const unsigned short* kh = (const unsigned short*)&ka;
    #pragma unroll
    for (int e = 0; e < 8; ++e) s += bf2f(qh[e]) * bf2f(kh[e]);
  }
  scores[idx] = s;
  atomicMin(&smin, f2key(s));
  __syncthreads();
  if (threadIdx.x == 0) atomicMin(minkey, smin);
}

// ---------------------------------------------------------------------------
// Attention phase 2: scale by 1/|min|, row L2-normalize, softmax over k,
// then out(8x64) = attn(8x8) @ V(8x64). One (b,h) per 64-thread block.
// ---------------------------------------------------------------------------
__global__ __launch_bounds__(64)
void attn_apply(const float* __restrict__ scores,
                const unsigned* __restrict__ minkey,
                const unsigned short* __restrict__ Vp,
                unsigned short* __restrict__ AttnOut) {
  __shared__ float P[64];
  int bh = blockIdx.x;
  int b = bh >> 4, h = bh & 15;
  int t = threadIdx.x;                                  // qi*8+ki
  float inv = 1.f / fabsf(key2f(*minkey));
  float s = scores[(size_t)bh * 64 + t] * inv;
  float ss = s * s;
  ss += __shfl_xor(ss, 1, 8); ss += __shfl_xor(ss, 2, 8); ss += __shfl_xor(ss, 4, 8);
  float v = s / fmaxf(sqrtf(ss), 1e-12f);
  float mx = v;
  mx = fmaxf(mx, __shfl_xor(mx, 1, 8));
  mx = fmaxf(mx, __shfl_xor(mx, 2, 8));
  mx = fmaxf(mx, __shfl_xor(mx, 4, 8));
  float e = __expf(v - mx);
  float sum = e;
  sum += __shfl_xor(sum, 1, 8); sum += __shfl_xor(sum, 2, 8); sum += __shfl_xor(sum, 4, 8);
  P[t] = e / sum;
  __syncthreads();
  int qi = t >> 3, dl = t & 7;
  float pr[8], o[8];
  #pragma unroll
  for (int k = 0; k < 8; ++k) pr[k] = P[qi * 8 + k];
  #pragma unroll
  for (int j = 0; j < 8; ++j) o[j] = 0.f;
  #pragma unroll
  for (int k = 0; k < 8; ++k) {
    const unsigned short* vr = Vp + (size_t)(b * 8 + k) * 1024 + h * 64;
    #pragma unroll
    for (int j = 0; j < 8; ++j) o[j] += pr[k] * bf2f(vr[dl + 8 * j]);
  }
  unsigned short* orow = AttnOut + (size_t)(b * 8 + qi) * 1024 + h * 64;
  #pragma unroll
  for (int j = 0; j < 8; ++j) orow[dl + 8 * j] = f2bf(o[j]);
}

// ---------------------------------------------------------------------------
// LayerNorm over d_model=1024, one row per 256-thread block.
// ---------------------------------------------------------------------------
__global__ __launch_bounds__(256)
void layernorm_k(const float* __restrict__ X, const float* __restrict__ g,
                 const float* __restrict__ bta, float* __restrict__ Y) {
  __shared__ float red[8];
  int row = blockIdx.x, t = threadIdx.x;
  const float* xr = X + (size_t)row * 1024;
  float4 xv = *(const float4*)(xr + t * 4);
  float s = xv.x + xv.y + xv.z + xv.w;
  #pragma unroll
  for (int d = 1; d < 32; d <<= 1) s += __shfl_xor(s, d, 32);
  if ((t & 31) == 0) red[t >> 5] = s;
  __syncthreads();
  float mean = (red[0]+red[1]+red[2]+red[3]+red[4]+red[5]+red[6]+red[7]) * (1.f/1024.f);
  __syncthreads();
  float d0 = xv.x - mean, d1 = xv.y - mean, d2 = xv.z - mean, d3 = xv.w - mean;
  float s2 = d0*d0 + d1*d1 + d2*d2 + d3*d3;
  #pragma unroll
  for (int d = 1; d < 32; d <<= 1) s2 += __shfl_xor(s2, d, 32);
  if ((t & 31) == 0) red[t >> 5] = s2;
  __syncthreads();
  float var = (red[0]+red[1]+red[2]+red[3]+red[4]+red[5]+red[6]+red[7]) * (1.f/1024.f);
  float inv = rsqrtf(var + 1e-10f);
  float4 gv = *(const float4*)(g + t * 4);
  float4 bv = *(const float4*)(bta + t * 4);
  float4 y;
  y.x = d0 * inv * gv.x + bv.x;  y.y = d1 * inv * gv.y + bv.y;
  y.z = d2 * inv * gv.z + bv.z;  y.w = d3 * inv * gv.w + bv.w;
  *(float4*)(Y + (size_t)row * 1024 + t * 4) = y;
}

// ---------------------------------------------------------------------------
// Fused ExternalAttention: per block = 2 batches (16 rows).
//   logits(16x64) = xln @ Wmk^T  (WMMA, K=1024)
//   softmax over modal axis (8 rows/batch), renormalize over S=64
//   q1(16x1024) = a @ Wmv^T      (WMMA, K=64), out = xln + q1
// ---------------------------------------------------------------------------
__global__ __launch_bounds__(256)
void ext_attn(const float* __restrict__ Xln,
              const unsigned short* __restrict__ Wmk,   // 64 x 1024 bf16
              const unsigned short* __restrict__ Wmv,   // 1024 x 64 bf16
              float* __restrict__ Out) {
  __shared__ unsigned short XA[16 * 1032];   // x_ln tile, bf16, 16B-mult stride
  __shared__ float LOG[16 * 64];
  __shared__ unsigned short AB[16 * 72];

  int t = threadIdx.x, lane = t & 31, wid = t >> 5;
  int l15 = lane & 15, kh = (lane >> 4) << 3, moff = (lane & 16) ? 8 : 0;
  size_t rowbase = (size_t)blockIdx.x * 16 * 1024;

  // 1) stage x_ln rows as bf16
  #pragma unroll
  for (int i = 0; i < 16; ++i) {
    int fl = t + 256 * i;                    // float4 id (4096 total)
    int row = fl >> 8, c4 = (fl & 255) << 2;
    float4 v = *(const float4*)(Xln + rowbase + (size_t)row * 1024 + c4);
    uint32_t p0 = (uint32_t)f2bf(v.x) | ((uint32_t)f2bf(v.y) << 16);
    uint32_t p1 = (uint32_t)f2bf(v.z) | ((uint32_t)f2bf(v.w) << 16);
    *(uint2*)&XA[row * 1032 + c4] = make_uint2(p0, p1);
  }
  __syncthreads();

  // 2) logits = XA @ Wmk^T (waves 0..3, one 16x16 N-tile each)
  if (wid < 4) {
    FragC acc;
    #pragma unroll
    for (int r = 0; r < 8; ++r) acc.f[r] = 0.f;
    int n = wid * 16 + l15;
    for (int kt = 0; kt < 1024; kt += 32) {
      FragAB a, bf;
      const unsigned short* ap = &XA[l15 * 1032 + kt + kh];
      a.u4[0] = *(const uint4*)ap;  a.u4[1] = *(const uint4*)(ap + 16);
      const unsigned short* bp = Wmk + (size_t)n * 1024 + kt + kh;
      bf.u4[0] = *(const uint4*)bp; bf.u4[1] = *(const uint4*)(bp + 16);
      acc.v = __builtin_amdgcn_wmma_f32_16x16x32_bf16(
          false, a.v, false, bf.v, (short)0, acc.v, false, false);
    }
    #pragma unroll
    for (int r = 0; r < 8; ++r) LOG[(moff + r) * 64 + wid * 16 + l15] = acc.f[r];
  }
  __syncthreads();

  // 3a) softmax over modal axis (within each batch half)
  if (t < 128) {
    int b2 = t >> 6, s = t & 63;
    float vv[8], mx = -3.0e38f;
    #pragma unroll
    for (int m = 0; m < 8; ++m) { vv[m] = LOG[(b2 * 8 + m) * 64 + s]; mx = fmaxf(mx, vv[m]); }
    float sum = 0.f;
    #pragma unroll
    for (int m = 0; m < 8; ++m) { vv[m] = __expf(vv[m] - mx); sum += vv[m]; }
    float invs = 1.f / sum;
    #pragma unroll
    for (int m = 0; m < 8; ++m) LOG[(b2 * 8 + m) * 64 + s] = vv[m] * invs;
  }
  __syncthreads();

  // 3b) renormalize over S, write bf16 A-matrix
  {
    int row = t >> 4, l = t & 15;
    float ps = 0.f;
    #pragma unroll
    for (int j = 0; j < 4; ++j) ps += LOG[row * 64 + l + 16 * j];
    ps += __shfl_xor(ps, 1, 16); ps += __shfl_xor(ps, 2, 16);
    ps += __shfl_xor(ps, 4, 16); ps += __shfl_xor(ps, 8, 16);
    float invr = 1.f / ps;
    #pragma unroll
    for (int j = 0; j < 4; ++j) {
      int s = l + 16 * j;
      AB[row * 72 + s] = f2bf(LOG[row * 64 + s] * invr);
    }
  }
  __syncthreads();

  // 4) q1 = a @ Wmv^T, out = x_ln + q1 (each wave: 16 x 128 N-strip)
  {
    FragC acc[8];
    #pragma unroll
    for (int g = 0; g < 8; ++g)
      #pragma unroll
      for (int r = 0; r < 8; ++r) acc[g].f[r] = 0.f;
    FragAB a[2];
    #pragma unroll
    for (int kt = 0; kt < 2; ++kt) {
      const unsigned short* ap = &AB[l15 * 72 + kt * 32 + kh];
      a[kt].u4[0] = *(const uint4*)ap;  a[kt].u4[1] = *(const uint4*)(ap + 16);
    }
    #pragma unroll
    for (int g = 0; g < 8; ++g) {
      int n = wid * 128 + g * 16 + l15;
      #pragma unroll
      for (int kt = 0; kt < 2; ++kt) {
        FragAB bf;
        const unsigned short* bp = Wmv + (size_t)n * 64 + kt * 32 + kh;
        bf.u4[0] = *(const uint4*)bp; bf.u4[1] = *(const uint4*)(bp + 16);
        acc[g].v = __builtin_amdgcn_wmma_f32_16x16x32_bf16(
            false, a[kt].v, false, bf.v, (short)0, acc[g].v, false, false);
      }
    }
    #pragma unroll
    for (int g = 0; g < 8; ++g) {
      int n = wid * 128 + g * 16 + l15;
      #pragma unroll
      for (int r = 0; r < 8; ++r) {
        size_t off = rowbase + (size_t)(moff + r) * 1024 + n;
        Out[off] = Xln[off] + acc[g].f[r];
      }
    }
  }
}

// ---------------------------------------------------------------------------
// Host-side orchestration
// ---------------------------------------------------------------------------
extern "C" void kernel_launch(void* const* d_in, const int* in_sizes, int n_in,
                              void* d_out, int out_size, void* d_ws, size_t ws_size,
                              hipStream_t stream) {
  (void)in_sizes; (void)n_in; (void)out_size; (void)ws_size;
  const float* q    = (const float*)d_in[0];
  const float* k    = (const float*)d_in[1];
  const float* v    = (const float*)d_in[2];
  const float* Wq   = (const float*)d_in[3];
  const float* Wk   = (const float*)d_in[4];
  const float* Wv   = (const float*)d_in[5];
  const float* Wfc  = (const float*)d_in[6];
  const float* gam  = (const float*)d_in[7];
  const float* bet  = (const float*)d_in[8];
  const float* Wmk  = (const float*)d_in[9];
  const float* Wmv  = (const float*)d_in[10];
  float* out = (float*)d_out;

  const int BS = 4096, D = 1024, S = 64, T = BS * 8;   // T = 32768 tokens

  char* ws = (char*)d_ws;
  size_t off = 0;
  auto alloc = [&](size_t bytes) {
    off = (off + 255) & ~(size_t)255;
    void* p = ws + off; off += bytes; return p;
  };
  unsigned short* wq_b  = (unsigned short*)alloc((size_t)D * D * 2);
  unsigned short* wk_b  = (unsigned short*)alloc((size_t)D * D * 2);
  unsigned short* wv_b  = (unsigned short*)alloc((size_t)D * D * 2);
  unsigned short* wfc_b = (unsigned short*)alloc((size_t)D * D * 2);
  unsigned short* wmk_b = (unsigned short*)alloc((size_t)S * D * 2);
  unsigned short* wmv_b = (unsigned short*)alloc((size_t)D * S * 2);
  unsigned short* Qp    = (unsigned short*)alloc((size_t)T * D * 2);  // 64MB
  unsigned short* Kp    = (unsigned short*)alloc((size_t)T * D * 2);  // 64MB
  unsigned short* Vp    = (unsigned short*)alloc((size_t)T * D * 2);  // 64MB
  unsigned short* Ao    = (unsigned short*)alloc((size_t)T * D * 2);  // 64MB
  float*    scores = (float*)alloc((size_t)BS * 16 * 64 * 4);         // 16MB
  unsigned* minkey = (unsigned*)alloc(256);
  // Overlays (regions are contiguous & 256B-aligned): fp32 FC output reuses
  // Qp+Kp (consumed by then); LayerNorm output reuses Vp+Ao.
  float* xfc = (float*)Qp;
  float* xln = (float*)Vp;

  // 1) weights -> bf16
  cvt_f32_to_bf16<<<(D * D + 255) / 256, 256, 0, stream>>>(Wq,  wq_b,  D * D);
  cvt_f32_to_bf16<<<(D * D + 255) / 256, 256, 0, stream>>>(Wk,  wk_b,  D * D);
  cvt_f32_to_bf16<<<(D * D + 255) / 256, 256, 0, stream>>>(Wv,  wv_b,  D * D);
  cvt_f32_to_bf16<<<(D * D + 255) / 256, 256, 0, stream>>>(Wfc, wfc_b, D * D);
  cvt_f32_to_bf16<<<(S * D + 255) / 256, 256, 0, stream>>>(Wmk, wmk_b, S * D);
  cvt_f32_to_bf16<<<(D * S + 255) / 256, 256, 0, stream>>>(Wmv, wmv_b, D * S);
  init_minkey<<<1, 1, 0, stream>>>(minkey);

  // 2) Q/K/V projections (1/sqrt(64) folded into Q)
  dim3 ggrid(D / 128, T / 128);
  gemm_tn_wmma<false, true, false><<<ggrid, 256, 0, stream>>>(q, wq_b, Qp, nullptr, T, D, D, 0.125f);
  gemm_tn_wmma<false, true, false><<<ggrid, 256, 0, stream>>>(k, wk_b, Kp, nullptr, T, D, D, 1.0f);
  gemm_tn_wmma<false, true, false><<<ggrid, 256, 0, stream>>>(v, wv_b, Vp, nullptr, T, D, D, 1.0f);

  // 3) attention scores + global min, then normalize/softmax/AV
  attn_scores<<<(BS * 16 * 64) / 256, 256, 0, stream>>>(Qp, Kp, scores, minkey);
  attn_apply<<<BS * 16, 64, 0, stream>>>(scores, minkey, Vp, Ao);

  // 4) FC + residual (fp32 out, overlays Qp/Kp; A staged via async-to-LDS)
  gemm_tn_wmma<true, false, true><<<ggrid, 256, 0, stream>>>(Ao, wfc_b, xfc, q, T, D, D, 1.0f);

  // 5) LayerNorm (overlays Vp/Ao)
  layernorm_k<<<T, 256, 0, stream>>>(xfc, gam, bet, xln);

  // 6) fused ExternalAttention + residual -> out
  ext_attn<<<BS / 2, 256, 0, stream>>>(xln, wmk_b, wmv_b, out);
}
